// EquiLinearLayer_58145267253823
// MI455X (gfx1250) — matrix operand order repaired and checked by reading, not compiled, not run on hardware
//
#include <hip/hip_runtime.h>

typedef __attribute__((ext_vector_type(16))) _Float16 v16h;
typedef __attribute__((ext_vector_type(8)))  float    v8f;
typedef __attribute__((ext_vector_type(4)))  float    v4f;

// Problem shape (fixed by reference)
#define M_TOK  32768   // B*S = 8*4096
#define N_DIM  1024    // OUT_CH*MV
#define K_DIM  1024    // IN_CH*MV
#define OUT_CH 64
#define IN_CH  64
#define NBLADE 9
#define MV     16

// ---------------------------------------------------------------------------
// Phase 1: Kmat[n=jy][k=ix] = sum_b W[j][i][b] * blade[b][x][y], emitted in
// f16, pre-swizzled into the WMMA B-fragment layout:
//   Bp[(((nt*32 + ks)*32 + lane)*16) + e] = Kmat[nt*16 + lane%16][ks*32 + (lane/16)*16 + e]
// Total 64*32*32*16 = 1,048,576 halves = 2 MB in d_ws.
// ---------------------------------------------------------------------------
__global__ void equi_build_B_f16(const float* __restrict__ W,
                                 const float* __restrict__ blade,
                                 _Float16* __restrict__ Bp) {
  int tid = blockIdx.x * blockDim.x + threadIdx.x;   // 0 .. 2^20-1
  int e  = tid & 15;
  int L  = (tid >> 4) & 31;
  int ks = (tid >> 9) & 31;
  int nt = tid >> 14;                                // 0..63
  int n = nt * 16 + (L & 15);                        // jy
  int k = ks * 32 + ((L >> 4) << 4) + e;             // ix
  int j = n >> 4, y = n & 15;
  int i = k >> 4, x = k & 15;
  float acc = 0.0f;
#pragma unroll
  for (int b = 0; b < NBLADE; ++b)
    acc += W[(j * IN_CH + i) * NBLADE + b] * blade[(b * MV + x) * MV + y];
  Bp[tid] = (_Float16)acc;
}

// ---------------------------------------------------------------------------
// Phase 2: C = A * Kmat^T with V_WMMA_F32_16X16X32_F16.
// One wave per 32(M) x 64(N) tile; 2x4 v8f accumulators; K loop = 32 steps.
// grid.x = N/64 = 16 (fastest) so the 16 waves sharing an A-tile are adjacent
// in dispatch order -> A tile read from HBM once, then L2/WGP$ hits.
// ---------------------------------------------------------------------------
__global__ void __launch_bounds__(32)
equi_gemm_wmma(const float* __restrict__ A,        // [M_TOK][K_DIM] fp32
               const _Float16* __restrict__ Bp,    // packed B fragments
               float* __restrict__ Out) {          // [M_TOK][N_DIM] fp32
  const int lane    = threadIdx.x;                 // 0..31
  const int ntBase  = blockIdx.x * 4;              // 4 N-subtiles of 16
  const int mBase   = blockIdx.y * 32;             // 2 M-subtiles of 16
  const int rowSub  = lane & 15;                   // M within subtile
  const int kLaneOf = (lane >> 4) * 16;            // A-frag per-lane K offset

  const v8f vzero = {0.f, 0.f, 0.f, 0.f, 0.f, 0.f, 0.f, 0.f};
  v8f acc[2][4];
#pragma unroll
  for (int ms = 0; ms < 2; ++ms)
#pragma unroll
    for (int ns = 0; ns < 4; ++ns) acc[ms][ns] = vzero;

  const float* Arow0 = A + (size_t)(mBase + rowSub) * K_DIM;
  const float* Arow1 = Arow0 + (size_t)16 * K_DIM;

  for (int kk = 0; kk < K_DIM / 32; ++kk) {
    const int kb = kk * 32;

    // --- A fragments: fp32 loads -> f16, contiguous k = 16*(lane/16) + e, as in Bp ---
    v16h af[2];
#pragma unroll
    for (int ms = 0; ms < 2; ++ms) {
      const float* ar = (ms ? Arow1 : Arow0) + kb + kLaneOf;
      v4f f0 = *(const v4f*)(ar +  0);   // K = c0 .. c0+3
      v4f f1 = *(const v4f*)(ar +  4);   // K = c0+4 .. c0+7
      v4f f2 = *(const v4f*)(ar +  8);   // K = c0+8 .. c0+11
      v4f f3 = *(const v4f*)(ar + 12);   // K = c0+12 .. c0+15
      v16h a;
#pragma unroll
      for (int t = 0; t < 4; ++t) {
        a[t]      = (_Float16)f0[t];
        a[4 + t]  = (_Float16)f1[t];
        a[8 + t]  = (_Float16)f2[t];
        a[12 + t] = (_Float16)f3[t];
      }
      af[ms] = a;
    }

    // --- B fragments (pre-packed, contiguous 32B per lane) + 8 WMMAs ---
#pragma unroll
    for (int ns = 0; ns < 4; ++ns) {
      const _Float16* bp =
          Bp + (((size_t)(ntBase + ns) * 32 + kk) * 32 + lane) * 16;
      v16h bf = *(const v16h*)bp;
#pragma unroll
      for (int ms = 0; ms < 2; ++ms)
        acc[ms][ns] = __builtin_amdgcn_wmma_f32_16x16x32_f16(
            false, af[ms], false, bf, (short)0, acc[ms][ns], false, false);
    }
  }

  // --- Store: D layout = lane -> N column, VGPR r -> row r + 8*(lane/16) ---
#pragma unroll
  for (int ms = 0; ms < 2; ++ms) {
    const int mRow = mBase + ms * 16 + (lane >> 4) * 8;
#pragma unroll
    for (int ns = 0; ns < 4; ++ns) {
      float* op = Out + (size_t)mRow * N_DIM + (size_t)(ntBase + ns) * 16 +
                  (lane & 15);
#pragma unroll
      for (int r = 0; r < 8; ++r) op[(size_t)r * N_DIM] = acc[ms][ns][r];
    }
  }
}

extern "C" void kernel_launch(void* const* d_in, const int* in_sizes, int n_in,
                              void* d_out, int out_size, void* d_ws,
                              size_t ws_size, hipStream_t stream) {
  const float* mv    = (const float*)d_in[0];  // [8,4096,64,16] fp32
  const float* W     = (const float*)d_in[1];  // [64,64,9] fp32
  const float* blade = (const float*)d_in[2];  // [9,16,16] fp32
  _Float16* Bp = (_Float16*)d_ws;              // 2 MB packed B fragments
  float* out = (float*)d_out;                  // [8,4096,64,16] fp32

  // Phase 1: 1M threads build the packed f16 kernel matrix.
  equi_build_B_f16<<<4096, 256, 0, stream>>>(W, blade, Bp);

  // Phase 2: token GEMM on the matrix cores.
  equi_gemm_wmma<<<dim3(N_DIM / 64, M_TOK / 32), 32, 0, stream>>>(mv, Bp, out);
}
